// PBE_13554916786510
// MI455X (gfx1250) — matrix operand order repaired and checked by reading, not compiled, not run on hardware
//
#include <hip/hip_runtime.h>
#include <math.h>

typedef __bf16 bf16;
typedef __attribute__((ext_vector_type(8)))  bf16  v8bf;
typedef __attribute__((ext_vector_type(16))) bf16  v16bf;
typedef __attribute__((ext_vector_type(8)))  float v8f;

#define DIMS 512
#define KNN  10

// Branch-guarded branch-free sorted insert: list ascending, l[KNN-1] = current max.
__device__ __forceinline__ void insert10(float (&l)[KNN], float d) {
  if (d < l[KNN - 1]) {
    float cur = d;
#pragma unroll
    for (int q = 0; q < KNN; ++q) {
      float mn = fminf(cur, l[q]);
      float mx = fmaxf(cur, l[q]);
      l[q] = mn;
      cur  = mx;
    }
  }
}

// Assemble a 16-element bf16 fragment from two contiguous 16-byte chunks.
__device__ __forceinline__ v16bf ldfrag(const bf16* p0, const bf16* p1) {
  v8bf a = *(const v8bf*)p0;
  v8bf b = *(const v8bf*)p1;
  return __builtin_shufflevector(a, b, 0, 1, 2, 3, 4, 5, 6, 7,
                                 8, 9, 10, 11, 12, 13, 14, 15);
}

// Pass 1: bf16 hi/lo split of X, plus per-row squared norms in fp32.
__global__ __launch_bounds__(128) void knn_prep(const float* __restrict__ X,
                                                bf16* __restrict__ Xhi,
                                                bf16* __restrict__ Xlo,
                                                float* __restrict__ sq) {
  const int row = blockIdx.x;
  const int tid = threadIdx.x;
  const float* xr = X + (size_t)row * DIMS;
  bf16* hr = Xhi + (size_t)row * DIMS;
  bf16* lr = Xlo + (size_t)row * DIMS;
  float s = 0.f;
#pragma unroll
  for (int e = tid; e < DIMS; e += 128) {
    float x = xr[e];
    bf16  h = (bf16)x;                 // RNE to bf16
    bf16  l = (bf16)(x - (float)h);    // residual
    hr[e] = h;
    lr[e] = l;
    s += x * x;
  }
  __shared__ float red[128];
  red[tid] = s;
  __syncthreads();
  for (int off = 64; off > 0; off >>= 1) {
    if (tid < off) red[tid] += red[tid + off];
    __syncthreads();
  }
  if (tid == 0) sq[row] = red[0];
}

// Pass 2: fused split-bf16 WMMA Gram tiles + per-row top-10 smallest distances.
__global__ __launch_bounds__(256) void knn_topk(const bf16* __restrict__ Xhi,
                                                const bf16* __restrict__ Xlo,
                                                const float* __restrict__ sq,
                                                float* __restrict__ out,
                                                int N) {
  __shared__ float cand[16][64][KNN];    // 40 KB: partial lists, 2 dump phases
  __shared__ float merged[16][16][KNN];  // 10 KB

  const int tid  = threadIdx.x;
  const int wave = tid >> 5;
  const int lane = tid & 31;
  const int l15  = lane & 15;
  const int lh   = lane >> 4;            // lane half: 0 or 1
  const int rowbase = blockIdx.x * 16;

  // Per-lane state: rows (rowbase + lh*8 + v), v = 0..7 (matches C/D VGPR layout).
  float lst[8][KNN];
#pragma unroll
  for (int v = 0; v < 8; ++v)
#pragma unroll
    for (int q = 0; q < KNN; ++q) lst[v][q] = __builtin_inff();

  float sqr[8];
#pragma unroll
  for (int v = 0; v < 8; ++v) sqr[v] = sq[rowbase + lh * 8 + v];

  // A fragment (16x32 bf16): lanes 0-15 row=lane, K {k0..k0+15};
  // lanes 16-31 same rows, K {k0+16..k0+31}.
  const bf16* ahi = Xhi + (size_t)(rowbase + l15) * DIMS + lh * 16;
  const bf16* alo = Xlo + (size_t)(rowbase + l15) * DIMS + lh * 16;

  const int ntiles = N >> 4;
  for (int jt = wave; jt < ntiles; jt += 8) {
    const int j0 = jt << 4;
    // B fragment (32x16 bf16): lane col = l15; lanes 0-15 K {k0..k0+15},
    // lanes 16-31 K {k0+16..k0+31} -> 32 contiguous bytes per lane.
    const bf16* bhi = Xhi + (size_t)(j0 + l15) * DIMS + lh * 16;
    const bf16* blo = Xlo + (size_t)(j0 + l15) * DIMS + lh * 16;

    v8f acc = {};
#pragma unroll 4
    for (int k0 = 0; k0 < DIMS; k0 += 32) {
      v16bf a_h = ldfrag(ahi + k0, ahi + k0 + 8);
      v16bf a_l = ldfrag(alo + k0, alo + k0 + 8);
      v16bf b_h = ldfrag(bhi + k0, bhi + k0 + 8);
      v16bf b_l = ldfrag(blo + k0, blo + k0 + 8);
      // split-bf16 fp32-accurate dot: hi*hi + hi*lo + lo*hi
      acc = __builtin_amdgcn_wmma_f32_16x16x32_bf16(false, a_h, false, b_h,
                                                    (short)0, acc, false, false);
      acc = __builtin_amdgcn_wmma_f32_16x16x32_bf16(false, a_h, false, b_l,
                                                    (short)0, acc, false, false);
      acc = __builtin_amdgcn_wmma_f32_16x16x32_bf16(false, a_l, false, b_h,
                                                    (short)0, acc, false, false);
    }

    const float sqc = sq[j0 + l15];
    const int   cg  = j0 + l15;
#pragma unroll
    for (int v = 0; v < 8; ++v) {
      float d2 = sqr[v] + sqc - 2.0f * acc[v];
      const int rg = rowbase + lh * 8 + v;
      d2 = (cg == rg) ? __builtin_inff() : d2;  // exclude self column
      insert10(lst[v], d2);
    }
  }

  // Merge 128 partial lists per row -> 16 per row, in two LDS-halving phases.
  float ml[KNN];
#pragma unroll
  for (int q = 0; q < KNN; ++q) ml[q] = __builtin_inff();
  const int mrow   = tid & 15;
  const int mchunk = tid >> 4;  // 0..15
  for (int p = 0; p < 2; ++p) {
    __syncthreads();
    if ((wave >> 2) == p) {
      const int src = (wave & 3) * 16 + l15;  // 0..63
#pragma unroll
      for (int v = 0; v < 8; ++v) {
        const int row = lh * 8 + v;
#pragma unroll
        for (int q = 0; q < KNN; ++q) cand[row][src][q] = lst[v][q];
      }
    }
    __syncthreads();
    for (int s = 0; s < 4; ++s)
#pragma unroll
      for (int q = 0; q < KNN; ++q) insert10(ml, cand[mrow][mchunk * 4 + s][q]);
  }
#pragma unroll
  for (int q = 0; q < KNN; ++q) merged[mrow][mchunk][q] = ml[q];
  __syncthreads();

  // Final per-row merge + log1p(mean(sqrt(.)))
  if (tid < 16) {
    const int row = tid;
    float fl[KNN];
#pragma unroll
    for (int q = 0; q < KNN; ++q) fl[q] = __builtin_inff();
    for (int c = 0; c < 16; ++c)
#pragma unroll
      for (int q = 0; q < KNN; ++q) insert10(fl, merged[row][c][q]);
    float ssum = 0.f;
#pragma unroll
    for (int q = 0; q < KNN; ++q) ssum += sqrtf(fmaxf(fl[q], 0.f));
    out[rowbase + row] = log1pf(ssum * (1.0f / KNN));
  }
}

extern "C" void kernel_launch(void* const* d_in, const int* in_sizes, int n_in,
                              void* d_out, int out_size, void* d_ws, size_t ws_size,
                              hipStream_t stream) {
  (void)n_in; (void)out_size; (void)ws_size;
  const float* X = (const float*)d_in[0];
  const int N = in_sizes[0] / DIMS;  // 16384

  // Workspace layout: Xhi (N*D bf16), Xlo (N*D bf16), sq (N f32) ~= 32.06 MB
  bf16*  Xhi = (bf16*)d_ws;
  bf16*  Xlo = Xhi + (size_t)N * DIMS;
  float* sqv = (float*)(Xlo + (size_t)N * DIMS);
  float* out = (float*)d_out;

  knn_prep<<<N, 128, 0, stream>>>(X, Xhi, Xlo, sqv);
  knn_topk<<<N / 16, 256, 0, stream>>>(Xhi, Xlo, sqv, out, N);
}
